// DecoderWithPositionLayer_188978561101
// MI455X (gfx1250) — matrix operand order repaired and checked by reading, not compiled, not run on hardware
//
#include <hip/hip_runtime.h>
#include <hip/hip_bf16.h>

// ---------------------------------------------------------------------------
// Types for CDNA5 WMMA (wave32, 16x16x32 f16 -> f32)
// ---------------------------------------------------------------------------
typedef _Float16 f16;
typedef __attribute__((ext_vector_type(8)))  _Float16 v8h;
typedef __attribute__((ext_vector_type(16))) _Float16 v16h;
typedef __attribute__((ext_vector_type(8)))  float    v8f;

#define HEADS 8
#define DH    128
#define BATCH 4
#define SEQN  1024
#define SEQM  1024
#define DMODEL 1024

// Async global->LDS (gfx1250): builtin signature is
//   (int4 addrspace(1)* gsrc, int4 addrspace(3)* ldst, imm offset, imm cpol)
#if defined(__has_builtin)
#if __has_builtin(__builtin_amdgcn_global_load_async_to_lds_b128)
#define HAVE_ASYNC_LDS 1
#endif
#endif

#ifdef HAVE_ASYNC_LDS
typedef int gv4i __attribute__((vector_size(16)));
typedef __attribute__((address_space(1))) gv4i as1_v4i;
typedef __attribute__((address_space(3))) gv4i as3_v4i;
#define ASYNC_LDS_B128(g, l)                                              \
  __builtin_amdgcn_global_load_async_to_lds_b128((as1_v4i*)(g),           \
                                                 (as3_v4i*)(l), 0, 0)
__device__ inline void wait_async_lds() {
#if __has_builtin(__builtin_amdgcn_s_wait_asynccnt)
  __builtin_amdgcn_s_wait_asynccnt(0);
#else
  asm volatile("s_wait_asynccnt 0" ::: "memory");
#endif
}
#endif

__device__ inline v16h frag_cat(v8h lo, v8h hi) {
  return __builtin_shufflevector(lo, hi, 0, 1, 2, 3, 4, 5, 6, 7, 8, 9, 10, 11,
                                 12, 13, 14, 15);
}

// A fragment (16x32 f16, M x K): lane<16 -> row=lane, K in {0..7,16..23};
// lane>=16 -> row=lane-16, K in {8..15,24..31}.  (ISA 7.12.2)
__device__ inline v16h load_a_frag(const f16* base, int stride, int lane) {
  int m  = lane & 15;
  int kb = (lane >> 4) * 8;
  const f16* p = base + m * stride + kb;
  v8h lo = *(const v8h*)(p);
  v8h hi = *(const v8h*)(p + 16);
  return frag_cat(lo, hi);
}

// B fragment (32x16 f16, K x N) stored N-major: lane<16 -> n=lane, K in {0..7,16..23};
// lane>=16 -> n=lane-16, K in {8..15,24..31}.
__device__ inline v16h load_b_frag(const f16* base, int stride, int lane) {
  int n  = lane & 15;
  int kb = (lane >> 4) * 8;
  const f16* p = base + n * stride + kb;
  v8h lo = *(const v8h*)(p);
  v8h hi = *(const v8h*)(p + 16);
  return frag_cat(lo, hi);
}

__device__ inline v8f wmma16(v16h a, v16h b, v8f c) {
  return __builtin_amdgcn_wmma_f32_16x16x32_f16(false, a, false, b, (short)0,
                                                c, false, false);
}

// ---------------------------------------------------------------------------
// Weight convert + transpose: W[k,n] f32 -> Wt[n,k] f16  (done once)
// ---------------------------------------------------------------------------
__global__ void convert_wt_kernel(const float* __restrict__ W,
                                  f16* __restrict__ Wt, int K, int N) {
  size_t idx = (size_t)blockIdx.x * 256 + threadIdx.x;
  if (idx >= (size_t)K * N) return;
  int k = (int)(idx / N);
  int n = (int)(idx % N);
  Wt[(size_t)n * K + k] = (f16)W[idx];
}

// ---------------------------------------------------------------------------
// LayerNorm (f32 in -> f16 out), one block per row, D = 1024
// ---------------------------------------------------------------------------
__global__ __launch_bounds__(256) void layernorm_kernel(
    const float* __restrict__ x, const float* __restrict__ g,
    const float* __restrict__ b, f16* __restrict__ out, int D) {
  __shared__ float red[256];
  int row = blockIdx.x, tid = threadIdx.x;
  const float* xr = x + (size_t)row * D;
  float v[4];
  float s = 0.f, s2 = 0.f;
#pragma unroll
  for (int i = 0; i < 4; i++) {
    float t = xr[tid + i * 256];
    v[i] = t;
    s += t;
    s2 += t * t;
  }
  red[tid] = s;
  __syncthreads();
  for (int off = 128; off > 0; off >>= 1) {
    if (tid < off) red[tid] += red[tid + off];
    __syncthreads();
  }
  float mean = red[0] / D;
  __syncthreads();
  red[tid] = s2;
  __syncthreads();
  for (int off = 128; off > 0; off >>= 1) {
    if (tid < off) red[tid] += red[tid + off];
    __syncthreads();
  }
  float var = red[0] / D - mean * mean;
  float inv = rsqrtf(var + 1e-3f);
#pragma unroll
  for (int i = 0; i < 4; i++) {
    int c = tid + i * 256;
    out[(size_t)row * D + c] = (f16)((v[i] - mean) * inv * g[c] + b[c]);
  }
}

// ---------------------------------------------------------------------------
// GEMM: C[M,N] = act(A[M,K] * Wt[N,K]^T + bias) (+ residual)
// A,Wt f16; out f32 and/or f16.  128x128 block tile, 8 waves (2x4),
// each wave 4x2 WMMA tiles, K-step 32, double-buffered async LDS pipeline.
// ---------------------------------------------------------------------------
#define GBM 128
#define GBN 128
#define GBK 32
#define LDT 40  // padded LDS k-stride (80B, multiple of 16B)

template <bool RELU>
__global__ __launch_bounds__(256) void gemm_f16_kernel(
    const f16* __restrict__ A, int lda, const f16* __restrict__ Bt, int ldb,
    const float* __restrict__ bias, const float* __restrict__ residual,
    int ldr, float* __restrict__ Cf, f16* __restrict__ Ch, int ldc, int K) {
  __shared__ f16 As[2][GBM * LDT];
  __shared__ f16 Bs[2][GBN * LDT];
  int tid = threadIdx.x;
  int lane = tid & 31, wave = tid >> 5;
  int wm = wave >> 2;  // 0..1 : 64-row half
  int wn = wave & 3;   // 0..3 : 32-col quarter
  int row0 = blockIdx.x * GBM, col0 = blockIdx.y * GBN;

  // stage K-tile kt into LDS buffer `buf` (async where available)
  auto stage = [&](int buf, int kt) {
#pragma unroll
    for (int p = 0; p < 2; p++) {
      int idx = tid + p * 256;
      int r = idx >> 2, c = idx & 3;
      const f16* ga = A + (size_t)(row0 + r) * lda + kt + c * 8;
      const f16* gb = Bt + (size_t)(col0 + r) * ldb + kt + c * 8;
      f16* la = &As[buf][r * LDT + c * 8];
      f16* lb = &Bs[buf][r * LDT + c * 8];
#ifdef HAVE_ASYNC_LDS
      ASYNC_LDS_B128(ga, la);
      ASYNC_LDS_B128(gb, lb);
#else
      *(v8h*)la = *(const v8h*)ga;
      *(v8h*)lb = *(const v8h*)gb;
#endif
    }
  };

  v8f acc[4][2] = {};

  stage(0, 0);  // prologue: fill buffer 0
  int cur = 0;
  for (int kt = 0; kt < K; kt += GBK) {
#ifdef HAVE_ASYNC_LDS
    wait_async_lds();  // loads issued one iteration ago (into buf `cur`)
#endif
    __syncthreads();   // buf `cur` published; all reads of buf `cur^1` retired
    if (kt + GBK < K) {
      stage(cur ^ 1, kt + GBK);  // prefetch next tile into other buffer
      __builtin_prefetch(A + (size_t)(row0 + (tid >> 1)) * lda + kt + GBK, 0, 3);
      __builtin_prefetch(Bt + (size_t)(col0 + (tid >> 1)) * ldb + kt + GBK, 0, 3);
    }
#pragma unroll
    for (int r = 0; r < 4; r++) {
      v16h a = load_a_frag(&As[cur][(wm * 64 + r * 16) * LDT], LDT, lane);
#pragma unroll
      for (int c = 0; c < 2; c++) {
        v16h b = load_b_frag(&Bs[cur][(wn * 32 + c * 16) * LDT], LDT, lane);
        acc[r][c] = wmma16(a, b, acc[r][c]);
      }
    }
    cur ^= 1;
  }

#pragma unroll
  for (int c = 0; c < 2; c++) {
    int col = col0 + wn * 32 + c * 16 + (lane & 15);
    float bv = bias ? bias[col] : 0.f;
#pragma unroll
    for (int r = 0; r < 4; r++) {
      int rbase = row0 + wm * 64 + r * 16 + 8 * (lane >> 4);
#pragma unroll
      for (int e = 0; e < 8; e++) {
        int row = rbase + e;
        float v = acc[r][c][e] + bv;
        if (RELU) v = fmaxf(v, 0.f);
        if (residual) v += residual[(size_t)row * ldr + col];
        if (Cf) Cf[(size_t)row * ldc + col] = v;
        if (Ch) Ch[(size_t)row * ldc + col] = (f16)v;
      }
    }
  }
}

// ---------------------------------------------------------------------------
// dots[b,h,i,r] = sum_d kk[b,i,h*DH+d] * pos_table[r, h*DH+d]
// grid (B*N, HEADS), 128 threads
// ---------------------------------------------------------------------------
__global__ __launch_bounds__(128) void dots_kernel(
    const f16* __restrict__ kk, int ldk, const float* __restrict__ pos_table,
    float* __restrict__ dots, int Nq) {
  __shared__ float red[128];
  int tid = threadIdx.x;
  int row = blockIdx.x;  // b*Nq + i
  int h = blockIdx.y;
  float kv = (float)kk[(size_t)row * ldk + h * DH + tid];
  int b = row / Nq, i = row % Nq;
#pragma unroll
  for (int r = 0; r < 3; r++) {
    red[tid] = kv * pos_table[r * DMODEL + h * DH + tid];
    __syncthreads();
    for (int off = 64; off > 0; off >>= 1) {
      if (tid < off) red[tid] += red[tid + off];
      __syncthreads();
    }
    if (tid == 0)
      dots[(((size_t)b * HEADS + h) * Nq + i) * 3 + r] = red[0];
    __syncthreads();
  }
}

// ---------------------------------------------------------------------------
// Flash-style attention, 64-row query block, dh=128.
// grid (nq/64, HEADS, B), 256 threads (8 waves).
// Wave w: row group rg=w>>1 (16 rows), dh half ch=w&1 (64 cols).
// S tiles: rows rg*16, cols ch*32 (2 WMMA tiles); O tiles: 4 per wave.
// ---------------------------------------------------------------------------
template <bool CAUSAL, bool HAS_BIAS>
__global__ __launch_bounds__(256) void attn_kernel(
    const f16* __restrict__ Q, int ldq, const f16* __restrict__ Km, int ldk,
    const f16* __restrict__ V, int ldv, float* __restrict__ O, int ldo,
    const unsigned char* __restrict__ vmask, const int* __restrict__ positions,
    const float* __restrict__ dots, int nq, int nk) {
  __shared__ f16 Ks[64 * DH];       // K block, N-major (j x d)
  __shared__ f16 Vt[DH * 64];       // V block transposed (d x j)
  __shared__ float Ssh[64 * 64];    // scores (f32); reused as P (f16)
  __shared__ float red_m[64][4], red_s[64][4];
  __shared__ float row_m[64], row_l[64], row_scale[64];

  int tid = threadIdx.x, lane = tid & 31, wave = tid >> 5;
  int rg = wave >> 1, ch = wave & 1;
  int ib = blockIdx.x, h = blockIdx.y, b = blockIdx.z;
  int i0 = ib * 64;
  const f16* Qb = Q + (size_t)b * nq * ldq + (size_t)h * DH;
  const f16* Kb = Km + (size_t)b * nk * ldk + (size_t)h * DH;
  const f16* Vb = V + (size_t)b * nk * ldv + (size_t)h * DH;

  // Q fragments for this wave's 16 rows, all 4 K-steps (kept in VGPRs)
  v16h qf[4];
  {
    int m = lane & 15, kb = (lane >> 4) * 8;
    const f16* qrow = Qb + (size_t)(i0 + rg * 16 + m) * ldq;
#pragma unroll
    for (int s = 0; s < 4; s++) {
      v8h lo = *(const v8h*)(qrow + s * 32 + kb);
      v8h hi = *(const v8h*)(qrow + s * 32 + kb + 16);
      qf[s] = frag_cat(lo, hi);
    }
  }
  for (int r = tid; r < 64; r += 256) { row_m[r] = -1e30f; row_l[r] = 0.f; }
  v8f oacc[4] = {};

  int jblocks = CAUSAL ? (ib + 1) : (nk / 64);
  for (int jb = 0; jb < jblocks; jb++) {
    int j0 = jb * 64;
    __syncthreads();  // previous iteration done reading Ks/Vt/P
    // stage K (async where available) and V transposed
#pragma unroll
    for (int p = 0; p < 4; p++) {
      int idx = tid + p * 256;
      int j = idx >> 4, c = idx & 15;
      const f16* gk = Kb + (size_t)(j0 + j) * ldk + c * 8;
#ifdef HAVE_ASYNC_LDS
      ASYNC_LDS_B128(gk, &Ks[j * DH + c * 8]);
#else
      *(v8h*)&Ks[j * DH + c * 8] = *(const v8h*)gk;
#endif
      v8h vv = *(const v8h*)(Vb + (size_t)(j0 + j) * ldv + c * 8);
#pragma unroll
      for (int u = 0; u < 8; u++) Vt[(c * 8 + u) * 64 + j] = vv[u];
    }
#ifdef HAVE_ASYNC_LDS
    wait_async_lds();
#endif
    __syncthreads();

    // S = Q K^T (this wave: rows rg*16, cols ch*32 .. +31)
    v8f s2[2] = {};
#pragma unroll
    for (int s = 0; s < 4; s++) {
#pragma unroll
      for (int c = 0; c < 2; c++) {
        v16h bf = load_b_frag(&Ks[(ch * 32 + c * 16) * DH + s * 32], DH, lane);
        s2[c] = wmma16(qf[s], bf, s2[c]);
      }
    }
    const float scale = 0.0883883476483f;  // 1/sqrt(128)
#pragma unroll
    for (int c = 0; c < 2; c++) {
      int coln = ch * 32 + c * 16 + (lane & 15);
#pragma unroll
      for (int e = 0; e < 8; e++) {
        int rown = rg * 16 + e + 8 * (lane >> 4);
        Ssh[rown * 64 + coln] = s2[c][e] * scale;
      }
    }
    __syncthreads();

    // online softmax: 4 threads per row, 16 cols each
    int row = tid >> 2, qq = tid & 3;
    int ig = i0 + row;
    float sv[16], lmax = -1e30f;
#pragma unroll
    for (int u = 0; u < 16; u++) {
      int col = qq * 16 + u, jg = j0 + col;
      float s = Ssh[row * 64 + col];
      if (HAS_BIAS) {
        int pr = positions[((size_t)b * nq + ig) * nq + jg];
        s += dots[(((size_t)b * HEADS + h) * nq + ig) * 3 + pr];
      }
      if (vmask && !vmask[(size_t)b * nk + jg]) s = -1e9f;
      if (CAUSAL && jg > ig) s = -1e9f;
      sv[u] = s;
      lmax = fmaxf(lmax, s);
    }
    red_m[row][qq] = lmax;
    __syncthreads();
    float bmax = fmaxf(fmaxf(red_m[row][0], red_m[row][1]),
                       fmaxf(red_m[row][2], red_m[row][3]));
    float m_old = row_m[row];
    float m_new = fmaxf(m_old, bmax);
    f16* P = (f16*)Ssh;  // safe: all Ssh reads captured in sv[] above
    float lsum = 0.f;
#pragma unroll
    for (int u = 0; u < 16; u++) {
      float e = __expf(sv[u] - m_new);
      lsum += e;
      P[row * 64 + qq * 16 + u] = (f16)e;
    }
    red_s[row][qq] = lsum;
    if (qq == 0) row_scale[row] = __expf(m_old - m_new);
    __syncthreads();
    if (qq == 0) {
      float ls = red_s[row][0] + red_s[row][1] + red_s[row][2] + red_s[row][3];
      row_l[row] = row_l[row] * row_scale[row] + ls;
      row_m[row] = m_new;
    }
    __syncthreads();

    // rescale accumulators, O += P @ V
    float sc[8];
#pragma unroll
    for (int e = 0; e < 8; e++)
      sc[e] = row_scale[rg * 16 + e + 8 * (lane >> 4)];
#pragma unroll
    for (int t = 0; t < 4; t++)
#pragma unroll
      for (int e = 0; e < 8; e++) oacc[t][e] *= sc[e];
#pragma unroll
    for (int s = 0; s < 2; s++) {
      v16h af;
      {
        int m = lane & 15, kb = (lane >> 4) * 8;
        const f16* pr = P + (rg * 16 + m) * 64 + s * 32;
        af = frag_cat(*(const v8h*)(pr + kb), *(const v8h*)(pr + kb + 16));
      }
#pragma unroll
      for (int t = 0; t < 4; t++) {
        v16h bf = load_b_frag(&Vt[(ch * 64 + t * 16) * 64 + s * 32], 64, lane);
        oacc[t] = wmma16(af, bf, oacc[t]);
      }
    }
  }
  __syncthreads();
  // normalize and store
#pragma unroll
  for (int t = 0; t < 4; t++) {
    int d = ch * 64 + t * 16 + (lane & 15);
#pragma unroll
    for (int e = 0; e < 8; e++) {
      int r = rg * 16 + e + 8 * (lane >> 4);
      float l = row_l[r];
      float v = (l > 0.f) ? oacc[t][e] / l : 0.f;
      O[((size_t)b * nq + i0 + r) * ldo + h * DH + d] = v;
    }
  }
}

// ---------------------------------------------------------------------------
// Host orchestration
// ---------------------------------------------------------------------------
extern "C" void kernel_launch(void* const* d_in, const int* in_sizes, int n_in,
                              void* d_out, int out_size, void* d_ws,
                              size_t ws_size, hipStream_t stream) {
  (void)in_sizes; (void)n_in; (void)out_size; (void)ws_size;
  const float* q_in    = (const float*)d_in[0];
  const float* v_enc   = (const float*)d_in[1];
  const unsigned char* q_mask = (const unsigned char*)d_in[2];
  const unsigned char* v_mask = (const unsigned char*)d_in[3];
  const int*   positions = (const int*)d_in[4];
  const float* pos_table = (const float*)d_in[5];
  const float* blk[5][6];
  for (int i = 0; i < 5; i++)
    for (int j = 0; j < 6; j++) blk[i][j] = (const float*)d_in[6 + i * 6 + j];
  // block param order: g, b, w1, b1, w2, b2

  const int R = BATCH * SEQN;  // 4096 rows everywhere
  const int w2n[5] = {3 * DMODEL, DMODEL, DMODEL, 2 * DMODEL, DMODEL};

  // ---- workspace carve ----
  char* ws = (char*)d_ws;
  size_t off = 0;
  auto carve = [&](size_t bytes) -> void* {
    void* p = ws + off;
    off += (bytes + 255) & ~(size_t)255;
    return p;
  };
  f16* wt1[5]; f16* wt2[5];
  for (int i = 0; i < 5; i++) {
    wt1[i] = (f16*)carve((size_t)512 * DMODEL * sizeof(f16));       // [512,1024]
    wt2[i] = (f16*)carve((size_t)w2n[i] * 512 * sizeof(f16));       // [n,512]
  }
  f16*   xn    = (f16*)carve((size_t)R * DMODEL * sizeof(f16));     // LN out
  f16*   hbuf  = (f16*)carve((size_t)R * 512 * sizeof(f16));        // hidden
  f16*   act0  = (f16*)carve((size_t)R * 3 * DMODEL * sizeof(f16)); // qq|kk|vv
  f16*   act3  = (f16*)carve((size_t)R * 2 * DMODEL * sizeof(f16)); // k2|v2
  f16*   q2buf = (f16*)carve((size_t)R * DMODEL * sizeof(f16));
  float* ybuf  = (float*)carve((size_t)R * DMODEL * sizeof(float));
  float* qcur  = (float*)carve((size_t)R * DMODEL * sizeof(float));
  float* dotsb = (float*)carve((size_t)BATCH * HEADS * SEQN * 3 * sizeof(float));

  // ---- convert + transpose weights to f16 N-major ----
  for (int i = 0; i < 5; i++) {
    size_t e1 = (size_t)DMODEL * 512;
    convert_wt_kernel<<<(unsigned)((e1 + 255) / 256), 256, 0, stream>>>(
        blk[i][2], wt1[i], DMODEL, 512);
    size_t e2 = (size_t)512 * w2n[i];
    convert_wt_kernel<<<(unsigned)((e2 + 255) / 256), 256, 0, stream>>>(
        blk[i][4], wt2[i], 512, w2n[i]);
  }

  dim3 gh(R / GBM, 512 / GBN);  // hidden GEMM grid (32,4)

  // ---- block 0: q -> act0 (qq|kk|vv) ----
  layernorm_kernel<<<R, 256, 0, stream>>>(q_in, blk[0][0], blk[0][1], xn, DMODEL);
  gemm_f16_kernel<true><<<gh, 256, 0, stream>>>(
      xn, DMODEL, wt1[0], DMODEL, blk[0][3], nullptr, 0, nullptr, hbuf, 512, DMODEL);
  gemm_f16_kernel<false><<<dim3(R / GBM, (3 * DMODEL) / GBN), 256, 0, stream>>>(
      hbuf, 512, wt2[0], 512, blk[0][5], nullptr, 0, nullptr, act0, 3 * DMODEL, 512);

  // ---- positional dots from kk ----
  dots_kernel<<<dim3(R, HEADS), 128, 0, stream>>>(
      act0 + DMODEL, 3 * DMODEL, pos_table, dotsb, SEQN);

  // ---- causal self-attention with bias ----
  attn_kernel<true, true><<<dim3(SEQN / 64, HEADS, BATCH), 256, 0, stream>>>(
      act0, 3 * DMODEL, act0 + DMODEL, 3 * DMODEL, act0 + 2 * DMODEL, 3 * DMODEL,
      ybuf, DMODEL, q_mask, positions, dotsb, SEQN, SEQN);

  // ---- block 1: qcur = q + block(y) ----
  layernorm_kernel<<<R, 256, 0, stream>>>(ybuf, blk[1][0], blk[1][1], xn, DMODEL);
  gemm_f16_kernel<true><<<gh, 256, 0, stream>>>(
      xn, DMODEL, wt1[1], DMODEL, blk[1][3], nullptr, 0, nullptr, hbuf, 512, DMODEL);
  gemm_f16_kernel<false><<<dim3(R / GBM, DMODEL / GBN), 256, 0, stream>>>(
      hbuf, 512, wt2[1], 512, blk[1][5], q_in, DMODEL, qcur, nullptr, DMODEL, 512);

  // ---- block 2: q2 = block(qcur) ----
  layernorm_kernel<<<R, 256, 0, stream>>>(qcur, blk[2][0], blk[2][1], xn, DMODEL);
  gemm_f16_kernel<true><<<gh, 256, 0, stream>>>(
      xn, DMODEL, wt1[2], DMODEL, blk[2][3], nullptr, 0, nullptr, hbuf, 512, DMODEL);
  gemm_f16_kernel<false><<<dim3(R / GBM, DMODEL / GBN), 256, 0, stream>>>(
      hbuf, 512, wt2[2], 512, blk[2][5], nullptr, 0, nullptr, q2buf, DMODEL, 512);

  // ---- block 3: k2|v2 = block(v_enc) ----
  layernorm_kernel<<<R, 256, 0, stream>>>(v_enc, blk[3][0], blk[3][1], xn, DMODEL);
  gemm_f16_kernel<true><<<gh, 256, 0, stream>>>(
      xn, DMODEL, wt1[3], DMODEL, blk[3][3], nullptr, 0, nullptr, hbuf, 512, DMODEL);
  gemm_f16_kernel<false><<<dim3(R / GBM, (2 * DMODEL) / GBN), 256, 0, stream>>>(
      hbuf, 512, wt2[3], 512, blk[3][5], nullptr, 0, nullptr, act3, 2 * DMODEL, 512);

  // ---- cross-attention (no bias, not causal) ----
  attn_kernel<false, false><<<dim3(SEQN / 64, HEADS, BATCH), 256, 0, stream>>>(
      q2buf, DMODEL, act3, 2 * DMODEL, act3 + DMODEL, 2 * DMODEL,
      ybuf, DMODEL, v_mask, nullptr, nullptr, SEQN, SEQM);

  // ---- block 4: out = qcur + block(y2) -> d_out ----
  layernorm_kernel<<<R, 256, 0, stream>>>(ybuf, blk[4][0], blk[4][1], xn, DMODEL);
  gemm_f16_kernel<true><<<gh, 256, 0, stream>>>(
      xn, DMODEL, wt1[4], DMODEL, blk[4][3], nullptr, 0, nullptr, hbuf, 512, DMODEL);
  gemm_f16_kernel<false><<<dim3(R / GBM, DMODEL / GBN), 256, 0, stream>>>(
      hbuf, 512, wt2[4], 512, blk[4][5], qcur, DMODEL, (float*)d_out, nullptr,
      DMODEL, 512);
}
